// CrossAttentionTransformer_12060268167855
// MI455X (gfx1250) — matrix operand order repaired and checked by reading, not compiled, not run on hardware
//
#include <hip/hip_runtime.h>
#include <hip/hip_bf16.h>
#include <math.h>

// ---------------------------------------------------------------------------
// CrossAttentionTransformer layer for MI455X (gfx1250, wave32, WMMA + TDM).
// GEMMs: v_wmma_f32_16x16x32_f16 (f16 in, f32 acc), A panel staged to LDS by
// the Tensor Data Mover once per block. Edge attention: f32 + global atomics.
// ---------------------------------------------------------------------------

typedef __attribute__((ext_vector_type(16))) _Float16 v16h;
typedef __attribute__((ext_vector_type(8)))  _Float16 v8h;
typedef __attribute__((ext_vector_type(4)))  _Float16 v4h;
typedef __attribute__((ext_vector_type(8)))  float    v8f;

#define D_DIM 128
#define D_MLP 512

// float <-> order-preserving unsigned encoding (for atomicMax on f32)
__device__ __forceinline__ unsigned fenc(float x) {
  unsigned b = __float_as_uint(x);
  return (b & 0x80000000u) ? ~b : (b | 0x80000000u);
}
__device__ __forceinline__ float fdec(unsigned e) {
  unsigned b = (e & 0x80000000u) ? (e & 0x7fffffffu) : ~e;
  return __uint_as_float(b);
}

// ---------------------- Tensor Data Mover panel load -----------------------
// Loads a contiguous [rows x cols] f16 panel (row stride == cols) from global
// memory into LDS at byte offset lds_off. 2D descriptor, groups 2/3 NULL-ish
// (zero), no multicast (workgroup_mask must be 0 outside clusters).
__device__ __forceinline__ void tdm_load_panel_f16(const _Float16* gptr,
                                                   unsigned lds_off,
                                                   unsigned rows, unsigned cols) {
#if defined(__gfx1250__) && __has_builtin(__builtin_amdgcn_tensor_load_to_lds)
  typedef __attribute__((ext_vector_type(4))) unsigned u32x4;
  typedef __attribute__((ext_vector_type(8))) int i32x8;
  typedef __attribute__((ext_vector_type(4))) int i32x4;
  unsigned long long ga = (unsigned long long)(size_t)gptr;
  u32x4 g0;
  g0[0] = 1u;                                        // count=1 (valid user D#)
  g0[1] = lds_off;                                   // lds_addr (bytes)
  g0[2] = (unsigned)(ga & 0xffffffffu);              // global_addr[31:0]
  g0[3] = (unsigned)((ga >> 32) & 0x01ffffffu)       // global_addr[56:32]
          | 0x80000000u;                             // type=2 ("image")
  i32x8 g1;
  g1[0] = 0x00010000;                                // data_size=1 (2B), mask=0
  g1[1] = (int)((cols & 0xffffu) << 16);             // tensor_dim0[15:0]
  g1[2] = (int)((cols >> 16) | ((rows & 0xffffu) << 16));   // dim0 hi | dim1 lo
  g1[3] = (int)(((rows >> 16) & 0xffffu) | ((cols & 0xffffu) << 16)); // dim1 hi | tile_dim0
  g1[4] = (int)(rows & 0xffffu);                     // tile_dim1=rows, tile_dim2=0
  g1[5] = (int)cols;                                 // tensor_dim0_stride[31:0]
  g1[6] = 0;                                         // stride0 hi, stride1 lo
  g1[7] = 0;
  i32x4 z4 = {0, 0, 0, 0};
#if __clang_major__ >= 23
  i32x8 z8 = {0, 0, 0, 0, 0, 0, 0, 0};
  __builtin_amdgcn_tensor_load_to_lds(g0, g1, z4, z4, z8, 0);
#else
  __builtin_amdgcn_tensor_load_to_lds(g0, g1, z4, z4, 0);
#endif
  __builtin_amdgcn_s_wait_tensorcnt(0);
#else
  (void)gptr; (void)lds_off; (void)rows; (void)cols;
#endif
}

// ---------------------------- utility kernels ------------------------------

__global__ __launch_bounds__(256) void k_fill_u32(unsigned* p, unsigned v, int n) {
  int i = blockIdx.x * 256 + threadIdx.x;
  if (i < n) p[i] = v;
}
__global__ __launch_bounds__(256) void k_fill_f32(float* p, float v, int n) {
  int i = blockIdx.x * 256 + threadIdx.x;
  if (i < n) p[i] = v;
}
__global__ __launch_bounds__(256) void k_copy_f32(const float* __restrict__ a,
                                                  float* __restrict__ b, int n) {
  int i = blockIdx.x * 256 + threadIdx.x;
  if (i < n) b[i] = a[i];
}
__global__ __launch_bounds__(256) void k_f32_to_f16(const float* __restrict__ a,
                                                    _Float16* __restrict__ b, int n) {
  int i = blockIdx.x * 256 + threadIdx.x;
  if (i < n) b[i] = (_Float16)a[i];
}
// W[K,N] f32 (row-major) -> Wt[N,K] f16 (row-major) so B fragments load contiguously
__global__ __launch_bounds__(256) void k_convW(const float* __restrict__ W,
                                               _Float16* __restrict__ Wt, int K, int N) {
  int i = blockIdx.x * 256 + threadIdx.x;
  if (i < K * N) {
    int kk = i / N, n = i - kk * N;
    Wt[(size_t)n * K + kk] = (_Float16)W[i];
  }
}

// LayerNorm over D=128, one wave per row, f16 output (feeds WMMA A operand).
__global__ __launch_bounds__(256)
void k_layernorm_f16(const float* __restrict__ X, const float* __restrict__ g,
                     const float* __restrict__ b, _Float16* __restrict__ Y, int rows) {
  int row = blockIdx.x * 8 + (threadIdx.x >> 5);
  if (row >= rows) return;
  int lane = threadIdx.x & 31;
  float4 x = ((const float4*)(X + (size_t)row * D_DIM))[lane];
  float s = x.x + x.y + x.z + x.w;
  for (int o = 16; o; o >>= 1) s += __shfl_xor(s, o, 32);
  float m = s * (1.0f / 128.0f);
  float dx = x.x - m, dy = x.y - m, dz = x.z - m, dw = x.w - m;
  float v = dx * dx + dy * dy + dz * dz + dw * dw;
  for (int o = 16; o; o >>= 1) v += __shfl_xor(v, o, 32);
  float rstd = rsqrtf(v * (1.0f / 128.0f) + 1e-5f);
  float4 gg = ((const float4*)g)[lane];
  float4 bb = ((const float4*)b)[lane];
  v4h y;
  y[0] = (_Float16)(dx * rstd * gg.x + bb.x);
  y[1] = (_Float16)(dy * rstd * gg.y + bb.y);
  y[2] = (_Float16)(dz * rstd * gg.z + bb.z);
  y[3] = (_Float16)(dw * rstd * gg.w + bb.w);
  *(v4h*)(Y + (size_t)row * D_DIM + lane * 4) = y;
}

// ---------------------------- WMMA GEMM ------------------------------------
// C[M,N] = act(A[M,K](f16) * Wt[N,K]^T(f16) + bias) (+ res).
// Block = 8 waves; block tile = 16 x 128. The 16xK A panel (contiguous) is
// DMA'd to LDS once by the TDM (wave 0), then the K loop is fully unrolled
// with no barriers: ds_load_b128 x2 + global_load_b128 x2 + wmma per step.
template <int KDIM, int ACT, int F16OUT, int RES>
__global__ __launch_bounds__(256)
void k_gemm_wmma(const _Float16* __restrict__ A, const _Float16* __restrict__ Wt,
                 const float* __restrict__ bias, const float* __restrict__ res,
                 void* __restrict__ Cout, int N) {
  __shared__ _Float16 sA[16 * KDIM];
  const int tid  = threadIdx.x;
  const int lane = tid & 31;
  const int wave = tid >> 5;
  const int m0 = blockIdx.y * 16;
  const int n0 = (blockIdx.x * 8 + wave) * 16;
  const int r  = lane & 15;      // row (A frag) / col (B frag, C col)
  const int hi = lane >> 4;      // half-wave select

#if defined(__gfx1250__) && __has_builtin(__builtin_amdgcn_tensor_load_to_lds)
  if (tid < 32) {   // one wave issues the DMA; TDM ignores EXEC
    tdm_load_panel_f16(A + (size_t)m0 * KDIM, (unsigned)(size_t)&sA[0], 16u,
                       (unsigned)KDIM);
  }
#else
  for (int i = tid; i < 16 * KDIM / 8; i += 256)
    ((v8h*)sA)[i] = ((const v8h*)(A + (size_t)m0 * KDIM))[i];
#endif
  __syncthreads();

  v8f acc = {};
#pragma unroll
  for (int k = 0; k < KDIM; k += 32) {
    // A fragment: lanes 0-15 rows M=0..15, K = {hi8..hi8+7, 16+hi8..16+hi8+7}
    const _Float16* ap = &sA[r * KDIM + k + hi * 8];
    v8h a0 = *(const v8h*)ap;
    v8h a1 = *(const v8h*)(ap + 16);
    // B fragment: lane = column, K = {hi8..hi8+7, 16+hi8..16+hi8+7}
    const _Float16* bp = &Wt[(size_t)(n0 + r) * KDIM + k + hi * 8];
    v8h b0 = *(const v8h*)bp;
    v8h b1 = *(const v8h*)(bp + 16);
    v16h af, bf;
#pragma unroll
    for (int i = 0; i < 8; ++i) {
      af[i] = a0[i]; af[i + 8] = a1[i];
      bf[i] = b0[i]; bf[i + 8] = b1[i];
    }
    acc = __builtin_amdgcn_wmma_f32_16x16x32_f16(false, af, false, bf,
                                                 (short)0, acc, false, false);
  }
  // epilogue: vgpr v -> row m0+v+8*hi, col n0+(lane&15)
  const int col = n0 + r;
  const float bc = bias[col];
#pragma unroll
  for (int v = 0; v < 8; ++v) {
    int row = m0 + v + 8 * hi;
    float x = acc[v] + bc;
    if (ACT) x = 0.5f * x * (1.0f + erff(x * 0.70710678118654752f));  // exact GELU
    if (RES) x += res[(size_t)row * N + col];
    if (F16OUT) ((_Float16*)Cout)[(size_t)row * N + col] = (_Float16)x;
    else        ((float*)Cout)[(size_t)row * N + col] = x;
  }
}

// ---------------------------- edge attention -------------------------------
// Pass 1: logits + segment max. One wave per edge (lane covers 4 of 128 dims).
__global__ __launch_bounds__(256)
void k_edge_logits(const float* __restrict__ Q, const float* __restrict__ K,
                   const int* __restrict__ qidx, const int* __restrict__ kidx,
                   const int* __restrict__ seg, float* __restrict__ elog,
                   unsigned* __restrict__ segmax, float scale, int E) {
  int e = blockIdx.x * 8 + (threadIdx.x >> 5);
  if (e >= E) return;
  int lane = threadIdx.x & 31;
  float4 q = ((const float4*)(Q + (size_t)qidx[e] * D_DIM))[lane];
  float4 k = ((const float4*)(K + (size_t)kidx[e] * D_DIM))[lane];
  float s = q.x * k.x + q.y * k.y + q.z * k.z + q.w * k.w;
  for (int o = 16; o; o >>= 1) s += __shfl_xor(s, o, 32);
  s *= scale;
  if (lane == 0) {
    elog[e] = s;
    atomicMax(&segmax[seg[e]], fenc(s));
  }
}

// Pass 2: exp(logit - max[seg]) and segment sum. One thread per edge.
__global__ __launch_bounds__(256)
void k_edge_exp(float* __restrict__ elog, const int* __restrict__ seg,
                const unsigned* __restrict__ segmax, float* __restrict__ segsum, int E) {
  int e = blockIdx.x * 256 + threadIdx.x;
  if (e >= E) return;
  int d = seg[e];
  float ex = __expf(elog[e] - fdec(segmax[d]));
  elog[e] = ex;
  atomicAdd(&segsum[d], ex);
}

// Pass 3: scatter a[e] * V[vidx[e]] into accum[seg[e]]. One wave per edge.
__global__ __launch_bounds__(256)
void k_edge_scatter(const float* __restrict__ elog, const int* __restrict__ seg,
                    const float* __restrict__ segsum, const float* __restrict__ V,
                    const int* __restrict__ vidx, float* __restrict__ accum, int E) {
  int e = blockIdx.x * 8 + (threadIdx.x >> 5);
  if (e >= E) return;
  int lane = threadIdx.x & 31;
  int d = seg[e];
  float coef = elog[e] / (segsum[d] + 1e-16f);
  float4 v = ((const float4*)(V + (size_t)vidx[e] * D_DIM))[lane];
  float* out = accum + (size_t)d * D_DIM + lane * 4;
  atomicAdd(out + 0, coef * v.x);
  atomicAdd(out + 1, coef * v.y);
  atomicAdd(out + 2, coef * v.z);
  atomicAdd(out + 3, coef * v.w);
}

// ---------------------------------------------------------------------------

extern "C" void kernel_launch(void* const* d_in, const int* in_sizes, int n_in,
                              void* d_out, int out_size, void* d_ws, size_t ws_size,
                              hipStream_t stream) {
  (void)n_in; (void)out_size; (void)ws_size;
  // ---- inputs (params flattened alphabetically, JAX-pytree order) ----
  const float* n0 = (const float*)d_in[0];
  const float* t0 = (const float*)d_in[1];
  const float* ln_nc_b = (const float*)d_in[2],  *ln_nc_g = (const float*)d_in[3];
  const float* ln_nm_b = (const float*)d_in[4],  *ln_nm_g = (const float*)d_in[5];
  const float* ln_ns_b = (const float*)d_in[6],  *ln_ns_g = (const float*)d_in[7];
  const float* ln_tc_b = (const float*)d_in[8],  *ln_tc_g = (const float*)d_in[9];
  const float* ln_tm_b = (const float*)d_in[10], *ln_tm_g = (const float*)d_in[11];
  const float* ln_ts_b = (const float*)d_in[12], *ln_ts_g = (const float*)d_in[13];
  const float* mn_fc1W = (const float*)d_in[14], *mn_fc1b = (const float*)d_in[15];
  const float* mn_fc2W = (const float*)d_in[16], *mn_fc2b = (const float*)d_in[17];
  const float* mt_fc1W = (const float*)d_in[18], *mt_fc1b = (const float*)d_in[19];
  const float* mt_fc2W = (const float*)d_in[20], *mt_fc2b = (const float*)d_in[21];
  const float* n2t_kW = (const float*)d_in[22],  *n2t_kb = (const float*)d_in[23];
  const float* n2t_oW = (const float*)d_in[24],  *n2t_ob = (const float*)d_in[25];
  const float* n2t_qW = (const float*)d_in[26],  *n2t_qb = (const float*)d_in[27];
  const float* n2t_vW = (const float*)d_in[28],  *n2t_vb = (const float*)d_in[29];
  const float* nsa_kW = (const float*)d_in[30],  *nsa_kb = (const float*)d_in[31];
  const float* nsa_qW = (const float*)d_in[32],  *nsa_qb = (const float*)d_in[33];
  const float* nsa_vW = (const float*)d_in[34],  *nsa_vb = (const float*)d_in[35];
  const float* t2n_kW = (const float*)d_in[36],  *t2n_kb = (const float*)d_in[37];
  const float* t2n_oW = (const float*)d_in[38],  *t2n_ob = (const float*)d_in[39];
  const float* t2n_qW = (const float*)d_in[40],  *t2n_qb = (const float*)d_in[41];
  const float* t2n_vW = (const float*)d_in[42],  *t2n_vb = (const float*)d_in[43];
  const float* tsa_kW = (const float*)d_in[44],  *tsa_kb = (const float*)d_in[45];
  const float* tsa_qW = (const float*)d_in[46],  *tsa_qb = (const float*)d_in[47];
  const float* tsa_vW = (const float*)d_in[48],  *tsa_vb = (const float*)d_in[49];
  const int* node_ei = (const int*)d_in[50];
  const int* tri_ei  = (const int*)d_in[51];
  const int* tn_ei   = (const int*)d_in[52];
  const int* nt_ei   = (const int*)d_in[53];

  const int N_NODE = in_sizes[0] / D_DIM;
  const int N_TRI  = in_sizes[1] / D_DIM;
  const int E_NODE = in_sizes[50] / 2;
  const int E_TRI  = in_sizes[51] / 2;
  const int E_TN   = in_sizes[52] / 2;
  const int E_NT   = in_sizes[53] / 2;
  const size_t NBe = (size_t)N_NODE * D_DIM;
  const size_t TBe = (size_t)N_TRI * D_DIM;
  const float scale = 0.088388347648318447f;  // 1/sqrt(128)

  // ---- workspace layout (bump allocator, 256B aligned) ----
  char* wsb = (char*)d_ws;
  size_t off = 0;
  auto alloc = [&](size_t bytes) -> void* {
    off = (off + 255) & ~(size_t)255;
    void* p = wsb + off;
    off += bytes;
    return p;
  };
  _Float16* lnNs16 = (_Float16*)alloc(NBe * 2);           // LN(n0, ln_ns) f16
  _Float16* lnTs16 = (_Float16*)alloc(TBe * 2);           // LN(t0, ln_ts) f16
  _Float16* wt     = (_Float16*)alloc((size_t)D_MLP * D_DIM * 2);  // weight f16^T
  float* Qb   = (float*)alloc(TBe * 4);
  float* Kb   = (float*)alloc(TBe * 4);
  float* Vb   = (float*)alloc(TBe * 4);
  float* aggr = (float*)alloc(TBe * 4);
  float* elog = (float*)alloc((size_t)E_NODE * 4);
  unsigned* segmax = (unsigned*)alloc((size_t)N_TRI * 4);
  float* segsum    = (float*)alloc((size_t)N_TRI * 4);
  float* x1    = (float*)alloc(TBe * 4);                  // node1 / tri1
  _Float16* x216 = (_Float16*)alloc(TBe * 2);             // rotating f16 buf
  _Float16* h16  = (_Float16*)alloc((size_t)N_TRI * D_MLP * 2);  // MLP hidden

  float* outNode = (float*)d_out;
  float* outTri  = (float*)d_out + NBe;

  // ---- host helpers ----
  auto ln = [&](const float* X, const float* g, const float* b, _Float16* Y, int rows) {
    k_layernorm_f16<<<(rows + 7) / 8, 256, 0, stream>>>(X, g, b, Y, rows);
  };
  auto gemm = [&](const _Float16* A, const float* W, const float* bias,
                  int M, int N, int K, int act_f16, const float* res, void* C) {
    k_convW<<<(K * N + 255) / 256, 256, 0, stream>>>(W, wt, K, N);
    dim3 grd(N / 128, M / 16);
    if (K == 512) {  // only mlp fc2: residual, f32 out
      k_gemm_wmma<512, 0, 0, 1><<<grd, 256, 0, stream>>>(A, wt, bias, res, C, N);
    } else if (act_f16) {
      k_gemm_wmma<128, 1, 1, 0><<<grd, 256, 0, stream>>>(A, wt, bias, nullptr, C, N);
    } else if (res) {
      k_gemm_wmma<128, 0, 0, 1><<<grd, 256, 0, stream>>>(A, wt, bias, res, C, N);
    } else {
      k_gemm_wmma<128, 0, 0, 0><<<grd, 256, 0, stream>>>(A, wt, bias, nullptr, C, N);
    }
  };
  auto attn = [&](const int* qidx, const int* kidx, const int* vidx, const int* seg,
                  int E, int Nseg, float* accum) {
    k_fill_u32<<<(Nseg + 255) / 256, 256, 0, stream>>>(segmax, 0x007FFFFFu, Nseg); // enc(-inf)
    k_fill_f32<<<(Nseg + 255) / 256, 256, 0, stream>>>(segsum, 0.0f, Nseg);
    k_edge_logits<<<(E + 7) / 8, 256, 0, stream>>>(Qb, Kb, qidx, kidx, seg, elog, segmax, scale, E);
    k_edge_exp<<<(E + 255) / 256, 256, 0, stream>>>(elog, seg, segmax, segsum, E);
    k_edge_scatter<<<(E + 7) / 8, 256, 0, stream>>>(elog, seg, segsum, Vb, vidx, accum, E);
  };

  // ---- shared LayerNorms of the block inputs ----
  ln(n0, ln_ns_g, ln_ns_b, lnNs16, N_NODE);
  ln(t0, ln_ts_g, ln_ts_b, lnTs16, N_TRI);

  // ========================== NODE branch ==========================
  // self-attention (Q from dst, K/V from src) over node_edge_index
  gemm(lnNs16, nsa_qW, nsa_qb, N_NODE, D_DIM, D_DIM, 0, nullptr, Qb);
  gemm(lnNs16, nsa_kW, nsa_kb, N_NODE, D_DIM, D_DIM, 0, nullptr, Kb);
  gemm(lnNs16, nsa_vW, nsa_vb, N_NODE, D_DIM, D_DIM, 0, nullptr, Vb);
  k_copy_f32<<<(int)((NBe + 255) / 256), 256, 0, stream>>>(n0, x1, (int)NBe);  // x1 = n0 (+attn)
  attn(node_ei + E_NODE, node_ei, node_ei, node_ei + E_NODE, E_NODE, N_NODE, x1);
  // cross-attention tri->node: q = lin(LN_ts(t0)[src]); k,v = lin(LN_nc(node1)[dst])
  ln(x1, ln_nc_g, ln_nc_b, x216, N_NODE);
  gemm(lnTs16, t2n_qW, t2n_qb, N_TRI, D_DIM, D_DIM, 0, nullptr, Qb);
  gemm(x216,   t2n_kW, t2n_kb, N_NODE, D_DIM, D_DIM, 0, nullptr, Kb);
  gemm(x216,   t2n_vW, t2n_vb, N_NODE, D_DIM, D_DIM, 0, nullptr, Vb);
  k_fill_f32<<<(int)((NBe + 255) / 256), 256, 0, stream>>>(aggr, 0.0f, (int)NBe);
  attn(tn_ei, tn_ei + E_TN, tn_ei + E_TN, tn_ei + E_TN, E_TN, N_NODE, aggr);
  k_f32_to_f16<<<(int)((NBe + 255) / 256), 256, 0, stream>>>(aggr, x216, (int)NBe);
  gemm(x216, t2n_oW, t2n_ob, N_NODE, D_DIM, D_DIM, 0, x1, outNode);   // node2 -> d_out
  // MLP with residual
  ln(outNode, ln_nm_g, ln_nm_b, x216, N_NODE);
  gemm(x216, mn_fc1W, mn_fc1b, N_NODE, D_MLP, D_DIM, 1, nullptr, h16);  // GELU, f16 out
  gemm(h16,  mn_fc2W, mn_fc2b, N_NODE, D_DIM, D_MLP, 0, outNode, outNode);

  // ========================== TRI branch ==========================
  gemm(lnTs16, tsa_qW, tsa_qb, N_TRI, D_DIM, D_DIM, 0, nullptr, Qb);
  gemm(lnTs16, tsa_kW, tsa_kb, N_TRI, D_DIM, D_DIM, 0, nullptr, Kb);
  gemm(lnTs16, tsa_vW, tsa_vb, N_TRI, D_DIM, D_DIM, 0, nullptr, Vb);
  k_copy_f32<<<(int)((TBe + 255) / 256), 256, 0, stream>>>(t0, x1, (int)TBe);
  attn(tri_ei + E_TRI, tri_ei, tri_ei, tri_ei + E_TRI, E_TRI, N_TRI, x1);
  // cross-attention node->tri
  ln(x1, ln_tc_g, ln_tc_b, x216, N_TRI);
  gemm(lnNs16, n2t_qW, n2t_qb, N_NODE, D_DIM, D_DIM, 0, nullptr, Qb);
  gemm(x216,   n2t_kW, n2t_kb, N_TRI, D_DIM, D_DIM, 0, nullptr, Kb);
  gemm(x216,   n2t_vW, n2t_vb, N_TRI, D_DIM, D_DIM, 0, nullptr, Vb);
  k_fill_f32<<<(int)((TBe + 255) / 256), 256, 0, stream>>>(aggr, 0.0f, (int)TBe);
  attn(nt_ei, nt_ei + E_NT, nt_ei + E_NT, nt_ei + E_NT, E_NT, N_TRI, aggr);
  k_f32_to_f16<<<(int)((TBe + 255) / 256), 256, 0, stream>>>(aggr, x216, (int)TBe);
  gemm(x216, n2t_oW, n2t_ob, N_TRI, D_DIM, D_DIM, 0, x1, outTri);     // tri2 -> d_out
  ln(outTri, ln_tm_g, ln_tm_b, x216, N_TRI);
  gemm(x216, mt_fc1W, mt_fc1b, N_TRI, D_MLP, D_DIM, 1, nullptr, h16);
  gemm(h16,  mt_fc2W, mt_fc2b, N_TRI, D_DIM, D_MLP, 0, outTri, outTri);
}
